// MultiHeadAttention_90099823936192
// MI455X (gfx1250) — matrix operand order repaired and checked by reading, not compiled, not run on hardware
//
#include <hip/hip_runtime.h>
#include <hip/hip_bf16.h>

typedef __bf16 bf16_t;
typedef __attribute__((ext_vector_type(16))) __bf16 v16bf;
typedef __attribute__((ext_vector_type(8)))  __bf16 v8bf;
typedef __attribute__((ext_vector_type(8)))  float  v8f;

#define T_ 2048
#define D_ 1024
#define H_ 16
#define DH_ 64

// ---------- workspace layout (bytes) ----------
#define OFF_CUM     (size_t)0                     // 2048 * 4
#define OFF_MISC    (size_t)8192                  // 2 ints (cum_last, done_any)
#define OFF_XB      (size_t)8448                  // 2048*1024 bf16
#define OFF_WQKVT   (size_t)(OFF_XB    + 4194304) // 3072*1024 bf16 (transposed)
#define OFF_WOUTT   (size_t)(OFF_WQKVT + 6291456) // 1024*1024 bf16 (transposed)
#define OFF_STATET  (size_t)(OFF_WOUTT + 2097152) // 16*64*64 bf16 (per-head transposed)
#define OFF_QB      (size_t)(OFF_STATET+ 131072)  // (H,T,DH) bf16
#define OFF_KB      (size_t)(OFF_QB    + 4194304) // (H,T,DH) bf16
#define OFF_KT      (size_t)(OFF_KB    + 4194304) // (H,DH,T) bf16, pre-masked by after_last
#define OFF_VT      (size_t)(OFF_KT    + 4194304) // (H,DH,T) bf16
#define OFF_OB      (size_t)(OFF_VT    + 4194304) // (T,D) bf16 attention output

__device__ __forceinline__ bf16_t f2bf(float f) {
    unsigned u = __builtin_bit_cast(unsigned, f);
    unsigned r = u + 0x7FFFu + ((u >> 16) & 1u);           // RNE
    return __builtin_bit_cast(bf16_t, (unsigned short)(r >> 16));
}

__device__ __forceinline__ v8f wmma_bf16(v16bf a, v16bf b, v8f c) {
    return __builtin_amdgcn_wmma_f32_16x16x32_bf16(false, a, false, b, (short)0, c,
                                                   false, false);
}

__device__ __forceinline__ v16bf cat16(v8bf lo, v8bf hi) {
    v16bf a;
#pragma unroll
    for (int i = 0; i < 8; ++i) { a[i] = lo[i]; a[8 + i] = hi[i]; }
    return a;
}

// ---------------- small prep kernels ----------------
__global__ void scan_kernel(const int* __restrict__ done, int* __restrict__ cum,
                            int* __restrict__ misc) {
    if (blockIdx.x == 0 && threadIdx.x == 0) {
        int s = 0;
        for (int t = 0; t < T_; ++t) { s += (done[t] != 0); cum[t] = s; }
        misc[0] = s;        // cum_last
        misc[1] = (s > 0);  // done_any
    }
}

__global__ void cvt_bf16_kernel(const float* __restrict__ in, bf16_t* __restrict__ out, int n) {
    int i = blockIdx.x * blockDim.x + threadIdx.x;
    if (i < n) out[i] = f2bf(in[i]);
}

// in: (B,R,C) f32 row-major -> out: (B,C,R) bf16 row-major
__global__ void transpose_bf16_kernel(const float* __restrict__ in, bf16_t* __restrict__ out,
                                      int B, int R, int C) {
    int i = blockIdx.x * blockDim.x + threadIdx.x;
    int total = B * R * C;
    if (i >= total) return;
    int b = i / (R * C);
    int r = (i / C) % R;
    int c = i % C;
    out[(size_t)b * R * C + (size_t)c * R + r] = f2bf(in[i]);
}

// ---------------- QKV projection GEMM (32x64 strip per wave) ----------------
// C(2048x3072) = xb(2048x1024) @ wqkv + bias ; epilogue scatters into q/k/v buffers.
__global__ __launch_bounds__(256) void gemm_qkv_kernel(
    const bf16_t* __restrict__ xb, const bf16_t* __restrict__ wT,
    const float* __restrict__ bias, const int* __restrict__ cum,
    const int* __restrict__ misc,
    bf16_t* __restrict__ qb, bf16_t* __restrict__ kb,
    bf16_t* __restrict__ kT, bf16_t* __restrict__ vT) {
    const int wave = (blockIdx.x * blockDim.x + threadIdx.x) >> 5;
    const int lane = threadIdx.x & 31;
    const int m = lane & 15, hi = lane >> 4;
    const int tr = wave / 48, tc = wave % 48;   // 64 x 48 strips of 32x64
    if (tr >= 64) return;
    const int t0 = tr * 32;
    const int c0 = tc * 64;

    const bf16_t* arow0 = xb + (size_t)(t0 + m) * D_;
    const bf16_t* arow1 = arow0 + (size_t)16 * D_;
    const bf16_t* brow[4];
#pragma unroll
    for (int nt = 0; nt < 4; ++nt) brow[nt] = wT + (size_t)(c0 + nt * 16 + m) * D_;

    v8f acc[2][4];
#pragma unroll
    for (int mt = 0; mt < 2; ++mt)
#pragma unroll
        for (int nt = 0; nt < 4; ++nt) acc[mt][nt] = (v8f){};

    for (int k0 = 0; k0 < D_; k0 += 32) {
        __builtin_prefetch(arow0 + k0 + 256, 0, 1);
        __builtin_prefetch(arow1 + k0 + 256, 0, 1);
        v16bf a0 = cat16(*(const v8bf*)(arow0 + k0 + hi * 8),
                         *(const v8bf*)(arow0 + k0 + 16 + hi * 8));
        v16bf a1 = cat16(*(const v8bf*)(arow1 + k0 + hi * 8),
                         *(const v8bf*)(arow1 + k0 + 16 + hi * 8));
#pragma unroll
        for (int nt = 0; nt < 4; ++nt) {
            v16bf b = cat16(*(const v8bf*)(brow[nt] + k0 + hi * 8),
                            *(const v8bf*)(brow[nt] + k0 + 16 + hi * 8));
            acc[0][nt] = wmma_bf16(a0, b, acc[0][nt]);
            acc[1][nt] = wmma_bf16(a1, b, acc[1][nt]);
        }
    }

    // whole strip lies in one (qkv, head) block: c0 % 64 == 0
    const int qkvi = c0 >> 10;
    const int h = (c0 >> 6) & 15;
    const int clast = misc[0];
    const bf16_t bz = f2bf(0.0f);
#pragma unroll
    for (int nt = 0; nt < 4; ++nt) {
        const int dh = nt * 16 + m;
        const float bsc = bias[c0 + nt * 16 + m];
#pragma unroll
        for (int mt = 0; mt < 2; ++mt) {
#pragma unroll
            for (int j = 0; j < 8; ++j) {
                const int t = t0 + mt * 16 + j + 8 * hi;
                bf16_t v = f2bf(acc[mt][nt][j] + bsc);
                if (qkvi == 0) {
                    qb[((size_t)(h * T_) + t) * DH_ + dh] = v;
                } else if (qkvi == 1) {
                    kb[((size_t)(h * T_) + t) * DH_ + dh] = v;
                    // pre-apply after_last mask: kT feeds only the state update
                    kT[((size_t)(h * DH_) + dh) * T_ + t] = (cum[t] == clast) ? v : bz;
                } else {
                    vT[((size_t)(h * DH_) + dh) * T_ + t] = v;
                }
            }
        }
    }
}

// ---------------- fused masked linear attention (32 t-rows per wave) ----------------
__global__ __launch_bounds__(256) void attn_kernel(
    const bf16_t* __restrict__ qb, const bf16_t* __restrict__ kb,
    const bf16_t* __restrict__ vT, const bf16_t* __restrict__ stateT,
    const int* __restrict__ cum, bf16_t* __restrict__ ob) {
    __shared__ __align__(32) bf16_t lds[8][2][16 * 40];
    const int wslot = threadIdx.x >> 5;
    const int wave = blockIdx.x * 8 + wslot;
    const int lane = threadIdx.x & 31;
    const int m = lane & 15, hi = lane >> 4;
    const int h = wave >> 6;          // 1024 waves = 16 heads * 64 t-blocks
    const int tb = wave & 63;
    const int t0 = tb * 32;

    // q A-fragments for both m-tiles, both d-chunks; reused across the whole s-loop
    v16bf aq[2][2];
#pragma unroll
    for (int mt = 0; mt < 2; ++mt) {
        const bf16_t* qrow = qb + ((size_t)(h * T_) + t0 + mt * 16 + m) * DH_;
        aq[mt][0] = cat16(*(const v8bf*)(qrow + hi * 16),
                          *(const v8bf*)(qrow + hi * 16 + 8));
        aq[mt][1] = cat16(*(const v8bf*)(qrow + 32 + hi * 16),
                          *(const v8bf*)(qrow + 32 + hi * 16 + 8));
    }
    int ct[2][8];
    bool bef[2];
#pragma unroll
    for (int mt = 0; mt < 2; ++mt) {
#pragma unroll
        for (int j = 0; j < 8; ++j) ct[mt][j] = cum[t0 + mt * 16 + j + 8 * hi];
        bef[mt] = (cum[t0 + mt * 16 + m] == 0);
    }

    v8f o[2][4];
#pragma unroll
    for (int mt = 0; mt < 2; ++mt)
#pragma unroll
        for (int nt = 0; nt < 4; ++nt) o[mt][nt] = (v8f){};

    // ---- carried-state term: O += (q * before_first) @ state[h] ----
    v16bf zero16;
#pragma unroll
    for (int i = 0; i < 16; ++i) zero16[i] = f2bf(0.0f);
    v16bf aqb[2][2];
#pragma unroll
    for (int mt = 0; mt < 2; ++mt) {
        aqb[mt][0] = bef[mt] ? aq[mt][0] : zero16;
        aqb[mt][1] = bef[mt] ? aq[mt][1] : zero16;
    }
#pragma unroll
    for (int nt = 0; nt < 4; ++nt) {
        const bf16_t* srow = stateT + ((size_t)(h * DH_) + nt * 16 + m) * DH_;
        v16bf b0 = *(const v16bf*)(srow + hi * 16);
        v16bf b1 = *(const v16bf*)(srow + 32 + hi * 16);
#pragma unroll
        for (int mt = 0; mt < 2; ++mt) {
            o[mt][nt] = wmma_bf16(aqb[mt][0], b0, o[mt][nt]);
            o[mt][nt] = wmma_bf16(aqb[mt][1], b1, o[mt][nt]);
        }
    }

    // ---- flash-style masked S @ v over causal s-chunks of 32 ----
    for (int sc = 0; sc <= tb; ++sc) {
        const int s0 = sc * 32;
        const bf16_t* k0row = kb + ((size_t)(h * T_) + s0 + m) * DH_;
        const bf16_t* k1row = k0row + 16 * DH_;
        v16bf b00 = *(const v16bf*)(k0row + hi * 16);
        v16bf b01 = *(const v16bf*)(k0row + 32 + hi * 16);
        v16bf b10 = *(const v16bf*)(k1row + hi * 16);
        v16bf b11 = *(const v16bf*)(k1row + 32 + hi * 16);
        const int sa = s0 + m, sb = s0 + 16 + m;
        const int csa = cum[sa], csb = cum[sb];

#pragma unroll
        for (int mt = 0; mt < 2; ++mt) {
            v8f sA = {}, sB = {};
            sA = wmma_bf16(aq[mt][0], b00, sA);
            sA = wmma_bf16(aq[mt][1], b01, sA);
            sB = wmma_bf16(aq[mt][0], b10, sB);
            sB = wmma_bf16(aq[mt][1], b11, sB);
            bf16_t* sl = &lds[wslot][mt][0];
#pragma unroll
            for (int j = 0; j < 8; ++j) {
                const int t = t0 + mt * 16 + j + 8 * hi;
                float v0 = (t >= sa && ct[mt][j] == csa) ? sA[j] : 0.0f;
                float v1 = (t >= sb && ct[mt][j] == csb) ? sB[j] : 0.0f;
                sl[(j + 8 * hi) * 40 + m]      = f2bf(v0);   // D-layout -> row-major
                sl[(j + 8 * hi) * 40 + 16 + m] = f2bf(v1);
            }
        }
        asm volatile("s_wait_dscnt 0" ::: "memory");          // wave-local LDS transpose
        v16bf as[2];
#pragma unroll
        for (int mt = 0; mt < 2; ++mt) {
            const bf16_t* sl = &lds[wslot][mt][0];
            as[mt] = cat16(*(const v8bf*)(sl + m * 40 + hi * 16),
                           *(const v8bf*)(sl + m * 40 + hi * 16 + 8));
        }
#pragma unroll
        for (int nt = 0; nt < 4; ++nt) {
            const bf16_t* vrow = vT + ((size_t)(h * DH_) + nt * 16 + m) * T_;
            v16bf bv = *(const v16bf*)(vrow + s0 + hi * 16);
#pragma unroll
            for (int mt = 0; mt < 2; ++mt)
                o[mt][nt] = wmma_bf16(as[mt], bv, o[mt][nt]);
        }
    }

    // epilogue: ob[t][h*64 + dh] (bf16) feeds the output projection GEMM
#pragma unroll
    for (int nt = 0; nt < 4; ++nt) {
        const int colc = h * DH_ + nt * 16 + m;
#pragma unroll
        for (int mt = 0; mt < 2; ++mt) {
#pragma unroll
            for (int j = 0; j < 8; ++j) {
                const int t = t0 + mt * 16 + j + 8 * hi;
                ob[(size_t)t * D_ + colc] = f2bf(o[mt][nt][j]);
            }
        }
    }
}

// ---------------- recurrent state update ----------------
// new_state[h] = (done_any ? 0 : state[h]) + k_aft^T @ v_aft ; kT is pre-masked.
__global__ __launch_bounds__(256) void state_kernel(
    const bf16_t* __restrict__ kT, const bf16_t* __restrict__ vT,
    const int* __restrict__ misc,
    const float* __restrict__ state_in, float* __restrict__ out_state) {
    const int wave = blockIdx.x * 8 + (threadIdx.x >> 5);
    if (wave >= 64) return;
    const int lane = threadIdx.x & 31;
    const int m = lane & 15, hi = lane >> 4;
    const int h = wave >> 2, ms = wave & 3;
    const int done_any = misc[1];

    const bf16_t* krow = kT + ((size_t)(h * DH_) + ms * 16 + m) * T_;
    v8f o[4];
#pragma unroll
    for (int nt = 0; nt < 4; ++nt) o[nt] = (v8f){};

    for (int t0 = 0; t0 < T_; t0 += 32) {
        v16bf a = cat16(*(const v8bf*)(krow + t0 + hi * 16),
                        *(const v8bf*)(krow + t0 + hi * 16 + 8));
#pragma unroll
        for (int nt = 0; nt < 4; ++nt) {
            const bf16_t* vrow = vT + ((size_t)(h * DH_) + nt * 16 + m) * T_;
            v16bf b = *(const v16bf*)(vrow + t0 + hi * 16);
            o[nt] = wmma_bf16(a, b, o[nt]);
        }
    }
#pragma unroll
    for (int nt = 0; nt < 4; ++nt) {
#pragma unroll
        for (int j = 0; j < 8; ++j) {
            const int M = ms * 16 + j + 8 * hi;   // d1
            const int N = nt * 16 + m;            // d2
            const size_t idx = (size_t)h * (DH_ * DH_) + (size_t)M * DH_ + N;
            float v = o[nt][j];
            if (!done_any) v += state_in[idx];
            out_state[idx] = v;
        }
    }
}

// ---------------- output projection GEMM (32x64 strip per wave) ----------------
__global__ __launch_bounds__(256) void gemm_out_kernel(
    const bf16_t* __restrict__ ob, const bf16_t* __restrict__ wT,
    const float* __restrict__ bias, float* __restrict__ xout) {
    const int wave = (blockIdx.x * blockDim.x + threadIdx.x) >> 5;
    const int lane = threadIdx.x & 31;
    const int m = lane & 15, hi = lane >> 4;
    const int tr = wave / 16, tc = wave % 16;   // 64 x 16 strips of 32x64
    if (tr >= 64) return;
    const int t0 = tr * 32;
    const int c0 = tc * 64;

    const bf16_t* arow0 = ob + (size_t)(t0 + m) * D_;
    const bf16_t* arow1 = arow0 + (size_t)16 * D_;
    const bf16_t* brow[4];
#pragma unroll
    for (int nt = 0; nt < 4; ++nt) brow[nt] = wT + (size_t)(c0 + nt * 16 + m) * D_;

    v8f acc[2][4];
#pragma unroll
    for (int mt = 0; mt < 2; ++mt)
#pragma unroll
        for (int nt = 0; nt < 4; ++nt) acc[mt][nt] = (v8f){};

    for (int k0 = 0; k0 < D_; k0 += 32) {
        __builtin_prefetch(arow0 + k0 + 256, 0, 1);
        __builtin_prefetch(arow1 + k0 + 256, 0, 1);
        v16bf a0 = cat16(*(const v8bf*)(arow0 + k0 + hi * 8),
                         *(const v8bf*)(arow0 + k0 + 16 + hi * 8));
        v16bf a1 = cat16(*(const v8bf*)(arow1 + k0 + hi * 8),
                         *(const v8bf*)(arow1 + k0 + 16 + hi * 8));
#pragma unroll
        for (int nt = 0; nt < 4; ++nt) {
            v16bf b = cat16(*(const v8bf*)(brow[nt] + k0 + hi * 8),
                            *(const v8bf*)(brow[nt] + k0 + 16 + hi * 8));
            acc[0][nt] = wmma_bf16(a0, b, acc[0][nt]);
            acc[1][nt] = wmma_bf16(a1, b, acc[1][nt]);
        }
    }
#pragma unroll
    for (int nt = 0; nt < 4; ++nt) {
        const int col = c0 + nt * 16 + m;
        const float bsc = bias[col];
#pragma unroll
        for (int mt = 0; mt < 2; ++mt) {
#pragma unroll
            for (int j = 0; j < 8; ++j) {
                const int t = t0 + mt * 16 + j + 8 * hi;
                xout[(size_t)t * D_ + col] = acc[mt][nt][j] + bsc;
            }
        }
    }
}

// ---------------- launch ----------------
extern "C" void kernel_launch(void* const* d_in, const int* in_sizes, int n_in,
                              void* d_out, int out_size, void* d_ws, size_t ws_size,
                              hipStream_t stream) {
    const float* state = (const float*)d_in[0];   // (16,64,64)
    const float* x     = (const float*)d_in[1];   // (2048,1024)
    const int*   done  = (const int*)d_in[2];     // (2048,)
    const float* w_qkv = (const float*)d_in[3];   // (1024,3072)
    const float* b_qkv = (const float*)d_in[4];   // (3072,)
    const float* w_out = (const float*)d_in[5];   // (1024,1024)
    const float* b_out = (const float*)d_in[6];   // (1024,)
    float* out = (float*)d_out;                   // [new_state(65536) | x_out(2097152)]

    char* ws = (char*)d_ws;
    int*    cum    = (int*)(ws + OFF_CUM);
    int*    misc   = (int*)(ws + OFF_MISC);
    bf16_t* xb     = (bf16_t*)(ws + OFF_XB);
    bf16_t* wqkvT  = (bf16_t*)(ws + OFF_WQKVT);
    bf16_t* woutT  = (bf16_t*)(ws + OFF_WOUTT);
    bf16_t* stateT = (bf16_t*)(ws + OFF_STATET);
    bf16_t* qb     = (bf16_t*)(ws + OFF_QB);
    bf16_t* kb     = (bf16_t*)(ws + OFF_KB);
    bf16_t* kT     = (bf16_t*)(ws + OFF_KT);
    bf16_t* vT     = (bf16_t*)(ws + OFF_VT);
    bf16_t* ob     = (bf16_t*)(ws + OFF_OB);

    scan_kernel<<<1, 1, 0, stream>>>(done, cum, misc);
    cvt_bf16_kernel<<<(T_ * D_ + 255) / 256, 256, 0, stream>>>(x, xb, T_ * D_);
    transpose_bf16_kernel<<<(1024 * 3072 + 255) / 256, 256, 0, stream>>>(w_qkv, wqkvT, 1, 1024, 3072);
    transpose_bf16_kernel<<<(1024 * 1024 + 255) / 256, 256, 0, stream>>>(w_out, woutT, 1, 1024, 1024);
    transpose_bf16_kernel<<<(H_ * DH_ * DH_ + 255) / 256, 256, 0, stream>>>(state, stateT, H_, DH_, DH_);

    gemm_qkv_kernel<<<384, 256, 0, stream>>>(xb, wqkvT, b_qkv, cum, misc,
                                             qb, kb, kT, vT);                 // 3072 strips
    attn_kernel<<<128, 256, 0, stream>>>(qb, kb, vT, stateT, cum, ob);        // 1024 waves
    state_kernel<<<8, 256, 0, stream>>>(kT, vT, misc, state, out);            // 64 waves
    gemm_out_kernel<<<128, 256, 0, stream>>>(ob, woutT, b_out, out + 65536);  // 1024 strips
}
